// FinalLayer_55525337202832
// MI455X (gfx1250) — matrix operand order repaired and checked by reading, not compiled, not run on hardware
//
#include <hip/hip_runtime.h>

typedef __attribute__((ext_vector_type(16))) __bf16        v16bf;
typedef __attribute__((ext_vector_type(8)))  float         v8f;
typedef __attribute__((ext_vector_type(8)))  unsigned int  v8u;
typedef __attribute__((ext_vector_type(4)))  unsigned int  v4u;
typedef __attribute__((ext_vector_type(8)))  int           v8i;
typedef __attribute__((ext_vector_type(4)))  int           v4i;

#define LN_EPS 1e-5f

static constexpr int M_ROWS = 8 * 2048;   // B*S = 16384
static constexpr int K_DIM  = 2048;       // D_IN
static constexpr int N_DIM  = 4096;       // D_OUT

static constexpr int BM = 128;
static constexpr int BN = 128;
static constexpr int BK = 32;

// LDS tile row stride: 16 dwords of data + 1 pad dword (TDM pad_interval=16DW, pad_amount=1DW)
static constexpr int TILE_STRIDE_U32 = BK / 2 + 1;   // 17

__device__ __forceinline__ unsigned short bf16h(float x) {
    unsigned int u = __builtin_bit_cast(unsigned int, x);
    u = (u + 0x7FFFu + ((u >> 16) & 1u)) >> 16;   // round-to-nearest-even
    return (unsigned short)u;
}
__device__ __forceinline__ unsigned int bf16pair(float lo, float hi) {
    return ((unsigned int)bf16h(hi) << 16) | (unsigned int)bf16h(lo);
}

__device__ __forceinline__ float fast_tanh(float x) {
#if __has_builtin(__builtin_amdgcn_tanhf)
    return __builtin_amdgcn_tanhf(x);     // v_tanh_f32 (TRANS32 on CDNA5)
#else
    return tanhf(x);
#endif
}

// ---- Tensor Data Mover: 2D bf16 tile [tile_rows x BK] -> LDS with per-row pad ----
__device__ __forceinline__ void tdm_load_2d(unsigned int lds_off, unsigned long long gaddr,
                                            unsigned int tensor_rows, unsigned int tile_rows) {
    v4u g0;
    g0[0] = 1u;                                                   // count=1, user descriptor
    g0[1] = lds_off;                                              // LDS byte address of tile start
    g0[2] = (unsigned int)gaddr;                                  // global_addr[31:0]
    g0[3] = (unsigned int)((gaddr >> 32) & 0x01FFFFFFull) | (2u << 30);  // addr[56:32] | type=2

    v8i g1;
    // data_size=1 (2B), pad_enable=1, pad_interval=3 (16 DWORDs), pad_amount=0 (1 DWORD)
    g1[0] = (int)((1u << 16) | (1u << 20) | (3u << 22));
    g1[1] = (int)(((unsigned)K_DIM & 0xFFFFu) << 16);             // tensor_dim0[15:0] = K
    g1[2] = (int)((tensor_rows & 0xFFFFu) << 16);                 // dim0 hi=0 | tensor_dim1[15:0]
    g1[3] = (int)(tensor_rows >> 16) | (int)((unsigned)BK << 16); // tensor_dim1 hi | tile_dim0=32
    g1[4] = (int)tile_rows;                                       // tile_dim1 | tile_dim2=0
    g1[5] = (int)K_DIM;                                           // tensor_dim0_stride = K elements
    g1[6] = 0;
    g1[7] = 0;

    v4i z = {};
#if __clang_major__ >= 23
    v8i z8 = {};
    __builtin_amdgcn_tensor_load_to_lds(g0, g1, z, z, z8, 0);
#else
    __builtin_amdgcn_tensor_load_to_lds(g0, g1, z, z, 0);
#endif
}

// ---------------- Kernel 1: fused LayerNorm -> packed bf16 rows [M][K] ----------------
__global__ void fl_ln_to_bf16(const float* __restrict__ x, unsigned int* __restrict__ abf) {
    __shared__ float s1[256];
    __shared__ float s2[256];
    __shared__ float mr[2];
    const int row = blockIdx.x;
    const int tid = threadIdx.x;
    const float* xr = x + (size_t)row * K_DIM;

    float sum = 0.f, sq = 0.f;
    #pragma unroll
    for (int j = 0; j < 8; ++j) {
        float v = xr[tid + j * 256];
        sum += v;
        sq  += v * v;
    }
    s1[tid] = sum;
    s2[tid] = sq;
    __syncthreads();
    for (int ofs = 128; ofs > 0; ofs >>= 1) {
        if (tid < ofs) { s1[tid] += s1[tid + ofs]; s2[tid] += s2[tid + ofs]; }
        __syncthreads();
    }
    if (tid == 0) {
        float mean = s1[0] * (1.0f / K_DIM);
        float var  = s2[0] * (1.0f / K_DIM) - mean * mean;
        mr[0] = mean;
        mr[1] = rsqrtf(var + LN_EPS);
    }
    __syncthreads();
    const float mean = mr[0], rstd = mr[1];

    const float2* xr2 = (const float2*)xr;
    unsigned int* orow = abf + (size_t)row * (K_DIM / 2);
    #pragma unroll
    for (int j = 0; j < 4; ++j) {
        int p = tid + j * 256;
        float2 ab = xr2[p];
        orow[p] = bf16pair((ab.x - mean) * rstd, (ab.y - mean) * rstd);
    }
}

// ---------------- Kernel 2: W[K][N] fp32 -> Wt[N][K] packed bf16 (transpose) ----------------
__global__ void fl_w_to_bf16t(const float* __restrict__ W, unsigned int* __restrict__ wt) {
    __shared__ unsigned short T[64][65];
    const int k0 = blockIdx.y * 64;
    const int n0 = blockIdx.x * 64;
    const int tid = threadIdx.x;

    #pragma unroll
    for (int j = 0; j < 16; ++j) {
        int idx = j * 256 + tid;          // 0..4095 over 64x64 tile
        int kk = idx >> 6;
        int nn = idx & 63;
        T[nn][kk] = bf16h(W[(size_t)(k0 + kk) * N_DIM + (n0 + nn)]);
    }
    __syncthreads();
    #pragma unroll
    for (int j = 0; j < 8; ++j) {
        int idx = j * 256 + tid;          // 0..2047
        int nn = idx >> 5;
        int kp = idx & 31;
        unsigned int lo = T[nn][2 * kp];
        unsigned int hi = T[nn][2 * kp + 1];
        wt[(size_t)(n0 + nn) * (K_DIM / 2) + (k0 >> 1) + kp] = lo | (hi << 16);
    }
}

// ------- Kernel 3: bf16 WMMA GEMM, TDM-staged double-buffered tiles, +bias, tanh -------
__global__ __launch_bounds__(256, 2) void fl_gemm_wmma(
    const unsigned int* __restrict__ Abf,   // [M][K/2] packed bf16 (normalized)
    const unsigned int* __restrict__ Wt,    // [N][K/2] packed bf16 (transposed W)
    const float* __restrict__ bias,
    float* __restrict__ out)
{
    __shared__ unsigned int As[2][BM * TILE_STRIDE_U32];   // TDM pads each 64B row with 4B
    __shared__ unsigned int Bs[2][BN * TILE_STRIDE_U32];

    const int tid  = threadIdx.x;
    const int lane = tid & 31;
    const int wave = tid >> 5;
    const int half = lane >> 4;
    const int l15  = lane & 15;
    const int wm   = wave >> 1;   // 0..3 : 32-row strip
    const int wn   = wave & 1;    // 0..1 : 64-col strip

    const int m0 = blockIdx.y * BM;
    const int n0 = blockIdx.x * BN;

    const unsigned int ldsA[2] = { (unsigned int)(size_t)&As[0][0], (unsigned int)(size_t)&As[1][0] };
    const unsigned int ldsB[2] = { (unsigned int)(size_t)&Bs[0][0], (unsigned int)(size_t)&Bs[1][0] };

    const unsigned long long aTile = (unsigned long long)(size_t)Abf + (size_t)m0 * (K_DIM * 2);
    const unsigned long long bTile = (unsigned long long)(size_t)Wt  + (size_t)n0 * (K_DIM * 2);

    v8f acc[2][4];
    {
        const v8f vzero = {};
        #pragma unroll
        for (int mi = 0; mi < 2; ++mi)
            #pragma unroll
            for (int ni = 0; ni < 4; ++ni)
                acc[mi][ni] = vzero;
    }

    // Prime the pipeline: buffer 0
    if (wave == 0) tdm_load_2d(ldsA[0], aTile, (unsigned)M_ROWS, (unsigned)BM);
    if (wave == 1) tdm_load_2d(ldsB[0], bTile, (unsigned)N_DIM, (unsigned)BN);

    int buf = 0;
    for (int k0 = 0; k0 < K_DIM; k0 += BK) {
        __syncthreads();   // previous reads of buf^1 are complete -> safe to overwrite it
        const bool has_next = (k0 + BK) < K_DIM;
        if (wave == 0) {
            if (has_next) {
                tdm_load_2d(ldsA[buf ^ 1], aTile + (unsigned long long)(k0 + BK) * 2,
                            (unsigned)M_ROWS, (unsigned)BM);
                __builtin_amdgcn_s_wait_tensorcnt(1);   // current tile done; next may be in flight
            } else {
                __builtin_amdgcn_s_wait_tensorcnt(0);
            }
        }
        if (wave == 1) {
            if (has_next) {
                tdm_load_2d(ldsB[buf ^ 1], bTile + (unsigned long long)(k0 + BK) * 2,
                            (unsigned)N_DIM, (unsigned)BN);
                __builtin_amdgcn_s_wait_tensorcnt(1);
            } else {
                __builtin_amdgcn_s_wait_tensorcnt(0);
            }
        }
        __syncthreads();   // buf is ready for everyone

        const unsigned int* Ab = &As[buf][0];
        const unsigned int* Bb = &Bs[buf][0];

        // ---- fragments per ISA 7.12.2 bf16 layouts (K-major rows, stride 17 uints) ----
        v16bf afrag[2];
        #pragma unroll
        for (int mi = 0; mi < 2; ++mi) {
            v8u au;
            const int m = wm * 32 + mi * 16 + l15;
            #pragma unroll
            for (int r = 0; r < 8; ++r) {
                // A 16x32 bf16: VGPR r holds K = {2(r&3), 2(r&3)+1} + 8*half + 16*(r>>2)
                const int kb = 2 * (r & 3) + 8 * half + 16 * (r >> 2);
                au[r] = Ab[m * TILE_STRIDE_U32 + (kb >> 1)];
            }
            afrag[mi] = __builtin_bit_cast(v16bf, au);
        }
        v16bf bfrag[4];
        #pragma unroll
        for (int ni = 0; ni < 4; ++ni) {
            v8u bu;
            const int n = wn * 64 + ni * 16 + l15;
            #pragma unroll
            for (int r = 0; r < 8; ++r) {
                // B 32x16 bf16: VGPR r holds K = {2(r&3), 2(r&3)+1} + 8*half + 16*(r>>2)
                const int kb = 2 * (r & 3) + 8 * half + 16 * (r >> 2);
                bu[r] = Bb[n * TILE_STRIDE_U32 + (kb >> 1)];
            }
            bfrag[ni] = __builtin_bit_cast(v16bf, bu);
        }

        #pragma unroll
        for (int mi = 0; mi < 2; ++mi)
            #pragma unroll
            for (int ni = 0; ni < 4; ++ni)
                acc[mi][ni] = __builtin_amdgcn_wmma_f32_16x16x32_bf16(
                    false, afrag[mi], false, bfrag[ni],
                    (short)0, acc[mi][ni], false, false);

        buf ^= 1;
    }

    // ---- epilogue: +bias, v_tanh_f32, fp32 store (C: VGPR r -> M=r+8*half, lane&15 -> N) ----
    #pragma unroll
    for (int ni = 0; ni < 4; ++ni) {
        const int n = n0 + wn * 64 + ni * 16 + l15;
        const float b = bias[n];
        #pragma unroll
        for (int mi = 0; mi < 2; ++mi) {
            const int mbase = m0 + wm * 32 + mi * 16 + 8 * half;
            #pragma unroll
            for (int r = 0; r < 8; ++r) {
                out[(size_t)(mbase + r) * N_DIM + n] = fast_tanh(acc[mi][ni][r] + b);
            }
        }
    }
}

extern "C" void kernel_launch(void* const* d_in, const int* in_sizes, int n_in,
                              void* d_out, int out_size, void* d_ws, size_t ws_size,
                              hipStream_t stream) {
    (void)in_sizes; (void)n_in; (void)out_size; (void)ws_size;
    const float* features = (const float*)d_in[0];   // (8, 2048, 2048) fp32
    const float* weight   = (const float*)d_in[1];   // (2048, 4096) fp32
    const float* biasp    = (const float*)d_in[2];   // (4096,) fp32
    float*       out      = (float*)d_out;           // (8, 2048, 4096) fp32

    // workspace: Abf (64 MB) | Wt (16 MB)
    unsigned int* Abf = (unsigned int*)d_ws;
    unsigned int* Wt  = (unsigned int*)((char*)d_ws + (size_t)M_ROWS * (K_DIM / 2) * sizeof(unsigned int));

    fl_ln_to_bf16<<<M_ROWS, 256, 0, stream>>>(features, Abf);
    fl_w_to_bf16t<<<dim3(N_DIM / 64, K_DIM / 64), 256, 0, stream>>>(weight, Wt);

    dim3 grid(N_DIM / BN, M_ROWS / BM);   // (32, 128)
    fl_gemm_wmma<<<grid, 256, 0, stream>>>(Abf, Wt, biasp, out);
}
